// PokeBNN_31610959298832
// MI455X (gfx1250) — matrix operand order repaired and checked by reading, not compiled, not run on hardware
//
#include <hip/hip_runtime.h>
#include <math.h>

#define CDIV(a,b) (((a)+(b)-1)/(b))

typedef __attribute__((ext_vector_type(16))) __bf16 v16bf;
typedef __attribute__((ext_vector_type(8)))  float  v8f;

// ---------------------------------------------------------------------------
// fp32 -> bf16 streaming convert
// ---------------------------------------------------------------------------
__global__ void k_f32_to_bf16(const float* __restrict__ in, __bf16* __restrict__ out, int n) {
  int i = blockIdx.x * blockDim.x + threadIdx.x;
  if (i < n) out[i] = (__bf16)in[i];
}

// HWIO fp32 [tap][CI][CO] -> bf16 [tap][CO][CI]  (B pre-transpose for WMMA)
__global__ void k_convert_w(const float* __restrict__ w, __bf16* __restrict__ out,
                            int taps, int CI, int CO) {
  int i = blockIdx.x * blockDim.x + threadIdx.x;
  int total = taps * CI * CO;
  if (i >= total) return;
  int ci = i % CI; int t = i / CI;
  int co = t % CO; int tap = t / CO;
  out[i] = (__bf16)w[((size_t)tap * CI + ci) * CO + co];
}

// ---------------------------------------------------------------------------
// Implicit-GEMM conv (1x1 or 3x3, stride 1/2, SAME) with fused BatchNorm.
// One wave32 computes a 64(M=pixels) x 64(CO) f32 tile: 4x4 register tiling,
// 16 independent v_wmma_f32_16x16x32_bf16 chains per 32-channel k-step.
//   A fragment (16x32 bf16): lane<16 -> K {0..7,16..23}; lane>=16 -> K {8..15,24..31}
//   B fragment (32x16 bf16): lane column = lane&15, K split as A ({0..7,16..23} | {8..15,24..31})
//   C/D: lane column = lane&15; rows M = (lane<16?0:8)+vgpr
// ---------------------------------------------------------------------------
__global__ void __launch_bounds__(32)
k_conv_wmma(const __bf16* __restrict__ xb, const __bf16* __restrict__ wb,
            float* __restrict__ y,
            const float* __restrict__ bs, const float* __restrict__ bb,
            const float* __restrict__ bm, const float* __restrict__ bv,
            int N, int IH, int IW, int CI,
            int OH, int OW, int CO, int KK, int stride, int padlo, int M) {
  const int lane   = threadIdx.x;
  const int r15    = lane & 15;
  const int mbase  = blockIdx.x * 64;
  const int cobase = blockIdx.y * 64;
  const int kab = (lane < 16) ? 0 : 8;    // A K-offset base
  const int kbb = (lane < 16) ? 0 : 8;    // B K-offset base
  const int ohw = OH * OW;

  // pixel coords of the 4 A-rows owned by this lane (clamped for ragged M)
  int nn[4], ohh[4], oww[4];
#pragma unroll
  for (int t = 0; t < 4; ++t) {
    int m = mbase + t * 16 + r15;
    if (m >= M) m = M - 1;
    int n = m / ohw; int rem = m - n * ohw;
    int oh = rem / OW;
    nn[t] = n; ohh[t] = oh; oww[t] = rem - oh * OW;
  }

  union U { v16bf v; float4 f[2]; };
  v8f zero = {};
  v8f acc[4][4];
#pragma unroll
  for (int t = 0; t < 4; ++t)
#pragma unroll
    for (int u = 0; u < 4; ++u) acc[t][u] = zero;

  const int taps = KK * KK;
  for (int tap = 0; tap < taps; ++tap) {
    const int ky = tap / KK, kx = tap - ky * KK;
    // per-tap A base pointers + in-bounds flags (hoisted out of k loop)
    const __bf16* abase[4];
    bool inb[4];
#pragma unroll
    for (int t = 0; t < 4; ++t) {
      const int ih = ohh[t] * stride - padlo + ky;
      const int iw = oww[t] * stride - padlo + kx;
      inb[t] = (ih >= 0) && (ih < IH) && (iw >= 0) && (iw < IW);
      abase[t] = xb + (((size_t)nn[t] * IH + ih) * IW + iw) * CI + kab;
    }
    const __bf16* wtap = wb + (size_t)tap * CO * CI;

    for (int ci0 = 0; ci0 < CI; ci0 += 32) {
      U B[4];
#pragma unroll
      for (int u = 0; u < 4; ++u) {
        const __bf16* brow = wtap + (size_t)(cobase + u * 16 + r15) * CI + ci0 + kbb;
        B[u].f[0] = *(const float4*)(brow);
        B[u].f[1] = *(const float4*)(brow + 16);
      }
      U A[4];
#pragma unroll
      for (int t = 0; t < 4; ++t) {
        if (inb[t]) {
          A[t].f[0] = *(const float4*)(abase[t] + ci0);
          A[t].f[1] = *(const float4*)(abase[t] + ci0 + 16);
        } else {
          A[t].f[0] = make_float4(0.f, 0.f, 0.f, 0.f);
          A[t].f[1] = make_float4(0.f, 0.f, 0.f, 0.f);
        }
      }
#pragma unroll
      for (int t = 0; t < 4; ++t)
#pragma unroll
        for (int u = 0; u < 4; ++u)
          acc[t][u] = __builtin_amdgcn_wmma_f32_16x16x32_bf16(
              false, A[t].v, false, B[u].v, (short)0, acc[t][u], false, false);
    }
  }

  // fused BN epilogue: one output channel per lane per u-tile
  const int mrow0 = (lane < 16) ? 0 : 8;
#pragma unroll
  for (int u = 0; u < 4; ++u) {
    const int co  = cobase + u * 16 + r15;
    const float inv = bs[co] / sqrtf(bv[co] + 1e-5f);
    const float sh  = bb[co] - bm[co] * inv;
#pragma unroll
    for (int t = 0; t < 4; ++t) {
#pragma unroll
      for (int rr = 0; rr < 8; ++rr) {
        const int mm = mbase + t * 16 + mrow0 + rr;
        if (mm < M) y[(size_t)mm * CO + co] = acc[t][u][rr] * inv + sh;
      }
    }
  }
}

// ---------------------------------------------------------------------------
// Init conv: 4x4 stride 4 VALID, 224x224x3 -> 56x56x32, fused bn + dprelu
// ---------------------------------------------------------------------------
__global__ void k_init_conv(const float* __restrict__ x, const float* __restrict__ w,
                            const float* bs, const float* bb, const float* bm, const float* bv,
                            const float* dbx, const float* dby, const float* dns, const float* dps,
                            float* __restrict__ out) {
  int i = blockIdx.x * blockDim.x + threadIdx.x;
  const int total = 16 * 56 * 56 * 32;
  if (i >= total) return;
  int co = i & 31; int t = i >> 5;
  int ow = t % 56; t /= 56;
  int oh = t % 56; int n = t / 56;
  float acc = 0.f;
  for (int ky = 0; ky < 4; ++ky)
    for (int kx = 0; kx < 4; ++kx)
      for (int ci = 0; ci < 3; ++ci)
        acc += x[(((size_t)n * 224 + (oh * 4 + ky)) * 224 + (ow * 4 + kx)) * 3 + ci]
             * w[((ky * 4 + kx) * 3 + ci) * 32 + co];
  float inv = bs[co] / sqrtf(bv[co] + 1e-5f);
  float v   = (acc - bm[co]) * inv + bb[co];
  float xb  = v - dbx[co];
  v = (xb >= 0.f ? dps[co] : dns[co]) * xb - dby[co];
  out[i] = v;
}

// ---------------------------------------------------------------------------
// Grouped conv: 3x3 SAME, 32 groups, 56x56x32 -> 56x56x64, fused bn + dprelu
// ---------------------------------------------------------------------------
__global__ void k_group_conv(const float* __restrict__ x, const float* __restrict__ w,
                             const float* bs, const float* bb, const float* bm, const float* bv,
                             const float* dbx, const float* dby, const float* dns, const float* dps,
                             float* __restrict__ out) {
  int i = blockIdx.x * blockDim.x + threadIdx.x;
  const int total = 16 * 56 * 56 * 64;
  if (i >= total) return;
  int co = i & 63; int t = i >> 6;
  int ow = t % 56; t /= 56;
  int oh = t % 56; int n = t / 56;
  int g = co >> 1;  // 64 outputs / 32 groups -> 2 per group; 1 input ch per group
  float acc = 0.f;
  for (int ky = 0; ky < 3; ++ky) {
    int ih = oh + ky - 1; if (ih < 0 || ih >= 56) continue;
    for (int kx = 0; kx < 3; ++kx) {
      int iw = ow + kx - 1; if (iw < 0 || iw >= 56) continue;
      acc += x[(((size_t)n * 56 + ih) * 56 + iw) * 32 + g] * w[(ky * 3 + kx) * 64 + co];
    }
  }
  float inv = bs[co] / sqrtf(bv[co] + 1e-5f);
  float v   = (acc - bm[co]) * inv + bb[co];
  float xb  = v - dbx[co];
  v = (xb >= 0.f ? dps[co] : dns[co]) * xb - dby[co];
  out[i] = v;
}

// ---------------------------------------------------------------------------
// SE: global average pool over H,W  -> [N,C]
// ---------------------------------------------------------------------------
__global__ void k_gap(const float* __restrict__ x, float* __restrict__ out,
                      int N, int H, int W, int C) {
  int i = blockIdx.x * blockDim.x + threadIdx.x;
  if (i >= N * C) return;
  int c = i % C, n = i / C;
  float acc = 0.f;
  const float* p = x + (size_t)n * H * W * C + c;
  for (int s = 0; s < H * W; ++s) acc += p[(size_t)s * C];
  out[i] = acc / (float)(H * W);
}

__global__ void k_fc_relu(const float* __restrict__ s0, const float* __restrict__ w,
                          const float* __restrict__ b, float* __restrict__ out,
                          int N, int CI, int J) {
  int i = blockIdx.x * blockDim.x + threadIdx.x;
  if (i >= N * J) return;
  int j = i % J, n = i / J;
  float acc = b[j];
  for (int ci = 0; ci < CI; ++ci) acc += s0[(size_t)n * CI + ci] * w[(size_t)ci * J + j];
  out[i] = acc > 0.f ? acc : 0.f;
}

__global__ void k_fc_hsig(const float* __restrict__ s1, const float* __restrict__ w,
                          const float* __restrict__ b, float* __restrict__ out,
                          int N, int J, int CO) {
  int i = blockIdx.x * blockDim.x + threadIdx.x;
  if (i >= N * CO) return;
  int co = i % CO, n = i / CO;
  float acc = b[co];
  for (int j = 0; j < J; ++j) acc += s1[(size_t)n * J + j] * w[(size_t)j * CO + co];
  acc += 3.f;
  acc = acc < 0.f ? 0.f : (acc > 6.f ? 6.f : acc);
  out[i] = acc * (1.f / 6.f);
}

// ---------------------------------------------------------------------------
// Residual fetch: channel fold-mean (nf) then optional 3x3/2 avg-pool (SAME,
// pad_lo = 0, sum/9 incl. zero padding)
// ---------------------------------------------------------------------------
__device__ __forceinline__
float res_term(const float* r, int n, int oh, int ow,
               int RH, int RW, int RC, int ch0, int nf, int pooled) {
  const float inv_nf = 1.f / (float)nf;
  if (!pooled) {
    const float* p = r + (((size_t)n * RH + oh) * RW + ow) * RC + ch0;
    float s = 0.f;
    for (int k = 0; k < nf; ++k) s += p[k];
    return s * inv_nf;
  }
  float acc = 0.f;
  for (int dy = 0; dy < 3; ++dy) {
    int ih = 2 * oh + dy; if (ih >= RH) continue;
    for (int dx = 0; dx < 3; ++dx) {
      int iw = 2 * ow + dx; if (iw >= RW) continue;
      const float* p = r + (((size_t)n * RH + ih) * RW + iw) * RC + ch0;
      float s = 0.f;
      for (int k = 0; k < nf; ++k) s += p[k];
      acc += s * inv_nf;
    }
  }
  return acc * (1.f / 9.f);
}

// ---------------------------------------------------------------------------
// Fused post: reshape_add(r,'zeropad') + reshape_add(r1,'tile') + dprelu
//             + SE gate + bn_split.  In-place on y.
// ---------------------------------------------------------------------------
__global__ void k_post(float* __restrict__ y, const float* __restrict__ r,
                       const float* __restrict__ r1, const float* __restrict__ se,
                       const float* dbx, const float* dby, const float* dns, const float* dps,
                       const float* bs, const float* bb, const float* bm, const float* bv,
                       int N, int OH, int OW, int CO,
                       int RC, int RH, int RW, int rpool,
                       int R1C, int R1H, int R1W, int r1pool) {
  int i = blockIdx.x * blockDim.x + threadIdx.x;
  if (i >= N * OH * OW * CO) return;
  int co = i % CO; int t = i / CO;
  int ow = t % OW; t /= OW;
  int oh = t % OH; int n = t / OH;

  float v = y[i];
  if (RC >= CO) {                 // fold-mean (ratio RC/CO, ==1 for identity)
    int nf = RC / CO;
    v += res_term(r, n, oh, ow, RH, RW, RC, co * nf, nf, rpool);
  } else if (co < RC) {           // zero-pad: only low channels get residual
    v += res_term(r, n, oh, ow, RH, RW, RC, co, 1, rpool);
  }
  if (r1) {                       // tile: channel co % R1C
    v += res_term(r1, n, oh, ow, R1H, R1W, R1C, co % R1C, 1, r1pool);
  }
  float xb = v - dbx[co];
  v = (xb >= 0.f ? dps[co] : dns[co]) * xb - dby[co];
  v *= se[(size_t)n * CO + co];
  float inv = bs[co] / sqrtf(bv[co] + 1e-5f);
  v = (v - bm[co]) * inv + bb[co];
  y[i] = v;
}

// ---------------------------------------------------------------------------
// Head: [N,C] x [C,1000]
// ---------------------------------------------------------------------------
__global__ void k_head(const float* __restrict__ g, const float* __restrict__ w,
                       float* __restrict__ out, int N, int C, int K) {
  int i = blockIdx.x * blockDim.x + threadIdx.x;
  if (i >= N * K) return;
  int k = i % K, n = i / K;
  float acc = 0.f;
  for (int c = 0; c < C; ++c) acc += g[(size_t)n * C + c] * w[(size_t)c * K + k];
  out[i] = acc;
}

// ---------------------------------------------------------------------------
extern "C" void kernel_launch(void* const* d_in, const int* in_sizes, int n_in,
                              void* d_out, int out_size, void* d_ws, size_t ws_size,
                              hipStream_t stream) {
  (void)in_sizes; (void)out_size; (void)ws_size;
  auto IN = [&](int i) -> const float* { return (const float*)d_in[i]; };
  const int N = 16;

  // -------- workspace carve (all chunks 256B-aligned sizes) --------
  char* p = (char*)d_ws;
  const size_t BIGE = 12845056;          // max activation elems (16*56*56*256)
  float* AB[3];
  for (int k = 0; k < 3; ++k) { AB[k] = (float*)p; p += BIGE * sizeof(float); }
  __bf16* XB = (__bf16*)p; p += BIGE * sizeof(__bf16);
  __bf16* WB = (__bf16*)p; p += (size_t)2359296 * sizeof(__bf16);  // max 9*512*512
  float* S0  = (float*)p;  p += (size_t)16 * 2048 * sizeof(float);
  float* S1  = (float*)p;  p += (size_t)16 * 256  * sizeof(float);
  float* SEB = (float*)p;  p += (size_t)16 * 2048 * sizeof(float);

  const dim3 b256(256);

  // -------- init stage --------
  {
    int total = N * 56 * 56 * 32;
    k_init_conv<<<CDIV(total, 256), b256, 0, stream>>>(
        IN(0), IN(1), IN(2), IN(3), IN(4), IN(5), IN(6), IN(7), IN(8), IN(9), AB[0]);
    total = N * 56 * 56 * 64;
    k_group_conv<<<CDIV(total, 256), b256, 0, stream>>>(
        AB[0], IN(10), IN(11), IN(12), IN(13), IN(14), IN(15), IN(16), IN(17), IN(18), AB[1]);
  }

  int xi = 1;
  int c = 64, H = 56, W = 56;
  static const int FEAT[16] = {64,64,64,128,128,128,128,256,256,256,256,256,256,512,512,512};
  static const int STR[16]  = {1, 1, 1, 2,  1,  1,  1,  2,  1,  1,  1,  1,  1,  2,  1,  1};

  for (int i = 0; i < 16; ++i) {
    const int s0i = xi;
    const int feat = FEAT[i], bstr = STR[i];
    float* r1 = AB[s0i];
    const int r1C = c, r1H = H, r1W = W;

    for (int j = 0; j < 3; ++j) {
      const int base = 19 + (i * 3 + j) * 17;
      const int kk = (j == 1) ? 3 : 1;
      const int st = (j == 1) ? bstr : 1;
      const int CO = (j == 2) ? feat * 4 : feat;
      const int CI = c;
      const int OH = H / st, OW = W / st;
      const int outIdx = (j == 1) ? (s0i + 2) % 3 : (s0i + 1) % 3;
      float* rb = AB[xi];
      float* yb = AB[outIdx];

      // pre-transpose + bf16-convert weights
      const int wtot = kk * kk * CI * CO;
      k_convert_w<<<CDIV(wtot, 256), b256, 0, stream>>>(IN(base + 0), WB, kk * kk, CI, CO);
      // bf16-convert activations
      const int xtot = N * H * W * CI;
      k_f32_to_bf16<<<CDIV(xtot, 256), b256, 0, stream>>>(rb, XB, xtot);

      // SE chain (on conv input rb)
      const int J = CI / 8;
      k_gap<<<CDIV(N * CI, 256), b256, 0, stream>>>(rb, S0, N, H, W, CI);
      k_fc_relu<<<CDIV(N * J, 256), b256, 0, stream>>>(S0, IN(base + 9), IN(base + 10), S1, N, CI, J);
      k_fc_hsig<<<CDIV(N * CO, 256), b256, 0, stream>>>(S1, IN(base + 11), IN(base + 12), SEB, N, J, CO);

      // conv + fused bn  (SAME padding: k3/s1 -> pad_lo 1; k3/s2 -> pad_lo 0)
      const int M = N * OH * OW;
      const int padlo = (kk == 3) ? ((st == 2) ? 0 : 1) : 0;
      dim3 grid(CDIV(M, 64), CO / 64);
      k_conv_wmma<<<grid, dim3(32), 0, stream>>>(
          XB, WB, yb, IN(base + 1), IN(base + 2), IN(base + 3), IN(base + 4),
          N, H, W, CI, OH, OW, CO, kk, st, padlo, M);

      // fused residual/dprelu/SE/bn_split (in-place on yb)
      const int ptot = M * CO;
      const float* r1p = (j == 2) ? r1 : nullptr;
      const int rpool  = (j == 1 && st == 2) ? 1 : 0;
      const int r1pool = (j == 2 && bstr == 2) ? 1 : 0;
      k_post<<<CDIV(ptot, 256), b256, 0, stream>>>(
          yb, rb, r1p, SEB,
          IN(base + 5), IN(base + 6), IN(base + 7), IN(base + 8),
          IN(base + 13), IN(base + 14), IN(base + 15), IN(base + 16),
          N, OH, OW, CO,
          CI, H, W, rpool,
          r1C, r1H, r1W, r1pool);

      c = CO; H = OH; W = OW; xi = outIdx;
    }
  }

  // -------- head: GAP (7x7x2048) + FC to 1000 --------
  k_gap<<<CDIV(N * 2048, 256), b256, 0, stream>>>(AB[xi], S0, N, 7, 7, 2048);
  k_head<<<CDIV(N * 1000, 256), b256, 0, stream>>>(S0, IN(n_in - 1), (float*)d_out, N, 2048, 1000);
}
